// RNNFromScratch_67259187855680
// MI455X (gfx1250) — compile-verified, matrix operand order checked by reading, not run on hardware
//
#include <hip/hip_runtime.h>
#include <hip/hip_bf16.h>

typedef _Float16 v16h __attribute__((ext_vector_type(16)));
typedef _Float16 v8h  __attribute__((ext_vector_type(8)));
typedef float    v8f  __attribute__((ext_vector_type(8)));
typedef float    v4f  __attribute__((ext_vector_type(4)));

#define EMB 256
#define HID 512
#define BATCH 128
#define SEQ 512
#define NTOK (BATCH * SEQ)
#define LN_EPS 1e-5f

// ---------------- Kernel 1: embedding gather + mask + LayerNorm -> f16 ----------------
// one wave (32 lanes) per token; each lane owns 8 of the 256 features.
__global__ void __launch_bounds__(256)
embed_ln_kernel(const int* __restrict__ x, const float* __restrict__ emb,
                const float* __restrict__ gamma, const float* __restrict__ beta,
                _Float16* __restrict__ e16) {
    int wave = (blockIdx.x * blockDim.x + threadIdx.x) >> 5;   // token id
    int lane = threadIdx.x & 31;
    if (wave >= NTOK) return;
    int xv = x[wave];
    float msk = (xv != 0) ? 1.0f : 0.0f;
    const v4f* rp = (const v4f*)(emb + xv * EMB + lane * 8);
    v4f a = rp[0], b = rp[1];
    float v[8];
    #pragma unroll
    for (int j = 0; j < 4; ++j) { v[j] = a[j] * msk; v[j + 4] = b[j] * msk; }
    float s = 0.f, sq = 0.f;
    #pragma unroll
    for (int j = 0; j < 8; ++j) { s += v[j]; sq += v[j] * v[j]; }
    #pragma unroll
    for (int o = 16; o >= 1; o >>= 1) {
        s  += __shfl_xor(s,  o, 32);
        sq += __shfl_xor(sq, o, 32);
    }
    float mu   = s * (1.0f / EMB);
    float var  = sq * (1.0f / EMB) - mu * mu;
    float rstd = rsqrtf(var + LN_EPS);
    v8h o16;
    #pragma unroll
    for (int j = 0; j < 8; ++j) {
        int idx = lane * 8 + j;
        o16[j] = (_Float16)((v[j] - mu) * rstd * gamma[idx] + beta[idx]);
    }
    *(v8h*)(e16 + wave * EMB + lane * 8) = o16;
}

// ---------------- Kernel 2: f32 -> f16 cast for weights ----------------
__global__ void cast_f32_f16(const float* __restrict__ in, _Float16* __restrict__ out, int n) {
    int i = blockIdx.x * blockDim.x + threadIdx.x;
    if (i < n) out[i] = (_Float16)in[i];
}

// ---------------- Kernel 3: input projection pre = e @ W_ih^T + b_ih ----------------
// WMMA tiling: one wave computes one 16x16 output tile, K-loop of 8 x wmma 16x16x32.
// A (16xK) lane layout: lane = m (0..15) + 16*hi; element e: vgpr v=e>>1,p=e&1;
//   k = (v<4 ? 0 : 16) + (v&3)*2 + p + 8*hi   (per ISA 16-bit A-matrix table)
// B (Kx16) mirrors A with lane -> column n; B[k][n] = W[n][k] (row-major weight rows).
__global__ void __launch_bounds__(256)
proj_kernel(const _Float16* __restrict__ e16, const _Float16* __restrict__ wih16,
            const float* __restrict__ bih, float* __restrict__ pre) {
    int gwave = (blockIdx.x * blockDim.x + threadIdx.x) >> 5;
    int lane  = threadIdx.x & 31;
    int mt = gwave >> 5;      // 0..4095  (65536/16 M-tiles)
    int nt = gwave & 31;      // 0..31    (512/16 N-tiles)
    int l15 = lane & 15, hi = lane >> 4;
    const _Float16* arow = e16   + (mt * 16 + l15) * EMB;
    const _Float16* brow = wih16 + (nt * 16 + l15) * EMB;
    v8f acc = {};
    #pragma unroll
    for (int kb = 0; kb < 8; ++kb) {
        int k0 = kb * 32 + hi * 8;
        v8h alo = *(const v8h*)(arow + k0);
        v8h ah2 = *(const v8h*)(arow + k0 + 16);
        v8h blo = *(const v8h*)(brow + k0);
        v8h bh2 = *(const v8h*)(brow + k0 + 16);
        v16h aa, bb;
        #pragma unroll
        for (int i = 0; i < 8; ++i) {
            aa[i] = alo[i]; aa[i + 8] = ah2[i];
            bb[i] = blo[i]; bb[i + 8] = bh2[i];
        }
        acc = __builtin_amdgcn_wmma_f32_16x16x32_f16(false, aa, false, bb,
                                                     (short)0, acc, false, false);
    }
    int n = nt * 16 + l15;
    float bn = bih[n];
    #pragma unroll
    for (int r = 0; r < 8; ++r) {          // C layout: row m = r + 8*hi
        int m = mt * 16 + r + 8 * hi;
        pre[m * HID + n] = acc[r] + bn;
    }
}

// ---------------- Kernel 4: recurrence h = tanh(pre_t + h @ W_hh^T + b_hh) ----------------
// 8 workgroups x 16 batch rows (independent recurrences -> no global sync).
// 32 waves/WG; wave w owns N columns [16w,16w+16); its W_hh slice lives in VGPRs.
// h kept as f16 in double-buffered LDS; one barrier per timestep.
__global__ void __launch_bounds__(1024)
rnn_kernel(const _Float16* __restrict__ whh16, const float* __restrict__ bhh,
           const float* __restrict__ pre, float* __restrict__ hfinal) {
    __shared__ __align__(16) _Float16 hsh[2][16][520];   // 520-stride: 16B-aligned rows, bank-spread
    int wave = threadIdx.x >> 5;           // 0..31 = N-tile
    int lane = threadIdx.x & 31;
    int l15 = lane & 15, hi = lane >> 4;
    int b0 = blockIdx.x * 16;
    int n  = wave * 16 + l15;

    // Preload this wave's 512x16 slice of W_hh^T as 16 B-fragments (register resident).
    v16h bfrag[16];
    const _Float16* brow = whh16 + n * HID;
    #pragma unroll
    for (int kb = 0; kb < 16; ++kb) {
        int k0 = kb * 32 + hi * 8;
        v8h blo = *(const v8h*)(brow + k0);
        v8h bh2 = *(const v8h*)(brow + k0 + 16);
        #pragma unroll
        for (int i = 0; i < 8; ++i) { bfrag[kb][i] = blo[i]; bfrag[kb][i + 8] = bh2[i]; }
    }
    float bn = bhh[n];

    // h0 = 0
    for (int i = threadIdx.x; i < 16 * 520; i += 1024) hsh[0][0][i] = (_Float16)0.f;
    __syncthreads();

    int buf = 0;
    for (int t = 0; t < SEQ; ++t) {
        // Prefetch pre[.,t,n] for this wave's 16x16 output tile (hides global latency
        // behind the WMMA chain).
        float pv[8];
        #pragma unroll
        for (int r = 0; r < 8; ++r) {
            int m = r + 8 * hi;
            pv[r] = pre[((b0 + m) * SEQ + t) * HID + n];
        }
        v8f acc = {};
        #pragma unroll
        for (int kb = 0; kb < 16; ++kb) {
            int k0 = kb * 32 + hi * 8;
            const _Float16* ar = &hsh[buf][l15][0];
            v8h alo = *(const v8h*)(ar + k0);          // ds_load_b128
            v8h ah2 = *(const v8h*)(ar + k0 + 16);
            v16h aa;
            #pragma unroll
            for (int i = 0; i < 8; ++i) { aa[i] = alo[i]; aa[i + 8] = ah2[i]; }
            acc = __builtin_amdgcn_wmma_f32_16x16x32_f16(false, aa, false, bfrag[kb],
                                                         (short)0, acc, false, false);
        }
        int nb = buf ^ 1;
        #pragma unroll
        for (int r = 0; r < 8; ++r) {
            int m = r + 8 * hi;
            float h = tanhf(acc[r] + pv[r] + bn);
            hsh[nb][m][n] = (_Float16)h;
            if (t == SEQ - 1) hfinal[(b0 + m) * HID + n] = h;
        }
        buf = nb;
        __syncthreads();   // writes of new h visible before next step's A-frag reads
    }
}

// ---------------- Kernel 5: out = h_last @ W_out^T + b_out  (128x4) ----------------
__global__ void out_kernel(const float* __restrict__ hfinal, const float* __restrict__ wout,
                           const float* __restrict__ bout, float* __restrict__ out) {
    int i = blockIdx.x * blockDim.x + threadIdx.x;
    if (i >= BATCH * 4) return;
    int b = i >> 2, o = i & 3;
    const float* hr = hfinal + b * HID;
    const float* wr = wout + o * HID;
    float s = bout[o];
    for (int k = 0; k < HID; ++k) s += hr[k] * wr[k];
    out[i] = s;
}

extern "C" void kernel_launch(void* const* d_in, const int* in_sizes, int n_in,
                              void* d_out, int out_size, void* d_ws, size_t ws_size,
                              hipStream_t stream) {
    const int*   x     = (const int*)d_in[0];
    const float* emb   = (const float*)d_in[1];
    const float* gamma = (const float*)d_in[2];
    const float* beta  = (const float*)d_in[3];
    const float* W_ih  = (const float*)d_in[4];
    const float* b_ih  = (const float*)d_in[5];
    const float* W_hh  = (const float*)d_in[6];
    const float* b_hh  = (const float*)d_in[7];
    const float* W_out = (const float*)d_in[8];
    const float* b_out = (const float*)d_in[9];
    float* out = (float*)d_out;

    char* ws = (char*)d_ws;
    _Float16* e16    = (_Float16*)(ws);                    // 65536*256*2  = 32 MB
    _Float16* wih16  = (_Float16*)(ws + 33554432);         // 512*256*2
    _Float16* whh16  = (_Float16*)(ws + 33816576);         // 512*512*2
    float*    pre    = (float*)(ws + 34340864);            // 65536*512*4  = 128 MB
    float*    hfin   = (float*)(ws + 168558592);           // 128*512*4

    embed_ln_kernel<<<NTOK / 8, 256, 0, stream>>>(x, emb, gamma, beta, e16);
    cast_f32_f16<<<(HID * EMB + 255) / 256, 256, 0, stream>>>(W_ih, wih16, HID * EMB);
    cast_f32_f16<<<(HID * HID + 255) / 256, 256, 0, stream>>>(W_hh, whh16, HID * HID);
    proj_kernel<<<(NTOK / 16) * (HID / 16) / 8, 256, 0, stream>>>(e16, wih16, b_ih, pre);
    rnn_kernel<<<BATCH / 16, 1024, 0, stream>>>(whh16, b_hh, pre, hfin);
    out_kernel<<<2, 256, 0, stream>>>(hfin, W_out, b_out, out);
}
